// CustomJacobiLayer_45243185496810
// MI455X (gfx1250) — matrix operand order repaired and checked by reading, not compiled, not run on hardware
//
#include <hip/hip_runtime.h>

typedef __attribute__((ext_vector_type(16))) __bf16 v16bf;
typedef __attribute__((ext_vector_type(8)))  __bf16 v8bf;
typedef __attribute__((ext_vector_type(8)))  float  v8f;

// Problem sizes (fixed by the reference)
constexpr int B_ = 4096;          // batch
constexpr int I_ = 512;           // input size
constexpr int D_ = 8;             // ORDER+1
constexpr int O_ = 512;           // output size
constexpr int K_ = I_ * D_;       // 4096 contraction length (k = i*8 + d)

// GEMM tiling
constexpr int BM  = 128;
constexpr int BN  = 128;
constexpr int BK  = 32;
constexpr int LDT = BK + 8;       // padded LDS row stride (elements): 80B rows, conflict-free b128
constexpr int NKT = K_ / BK;      // 128 K-steps

#if defined(__has_builtin)
#if __has_builtin(__builtin_amdgcn_tensor_load_to_lds)
#define CDNA5_HAS_TDM 1
#endif
#endif
#ifndef CDNA5_HAS_TDM
#define CDNA5_HAS_TDM 0
#endif

// ---------------- phase 0: coef f32 -> bf16 (layout already [O, K] row-major) ----------------
__global__ void coef_to_bf16_kernel(const float* __restrict__ coef, __bf16* __restrict__ out) {
  int t = blockIdx.x * blockDim.x + threadIdx.x;       // one thread per 8 elements
  const float4* p = (const float4*)coef + (size_t)t * 2;
  float4 a = p[0], b = p[1];
  v8bf r;
  r[0] = (__bf16)a.x; r[1] = (__bf16)a.y; r[2] = (__bf16)a.z; r[3] = (__bf16)a.w;
  r[4] = (__bf16)b.x; r[5] = (__bf16)b.y; r[6] = (__bf16)b.z; r[7] = (__bf16)b.w;
  ((v8bf*)out)[t] = r;
}

// ---------------- phase 1: u = tanh(x); Jacobi(alpha=beta=1) recurrence; pack 8 x bf16 --------
__global__ void jacobi_bf16_kernel(const float* __restrict__ x, __bf16* __restrict__ jac) {
  int t = blockIdx.x * blockDim.x + threadIdx.x;       // one thread per (b, i)
  float u = tanhf(x[t]);
  // alpha=beta=1: K2[i]=0; K1[i] = (2i+2)(2i+1)/(2i(i+2)),  K3[i] = (i+1)/(i+2)
  const float k1[8] = {0.f, 0.f, 1.875f, 1.86666667f, 1.875f, 1.88571429f, 1.89583333f, 1.90476190f};
  const float k3[8] = {0.f, 0.f, 0.75f,  0.8f,        0.83333333f, 0.85714286f, 0.875f, 0.88888889f};
  float p[8];
  p[0] = 1.0f;
  p[1] = 2.0f * u;                 // 0.5*(a+b+2)*u - 0.5*(a-b) with a=b=1
#pragma unroll
  for (int i = 2; i < 8; ++i) p[i] = k1[i] * u * p[i - 1] - k3[i] * p[i - 2];
  v8bf r;
#pragma unroll
  for (int i = 0; i < 8; ++i) r[i] = (__bf16)p[i];
  ((v8bf*)jac)[t] = r;             // jac[b*K + i*8 + d] -- matches coef's k ordering
}

#if CDNA5_HAS_TDM
typedef __attribute__((ext_vector_type(4))) unsigned int u32x4;
typedef __attribute__((ext_vector_type(4))) int          i32x4;
typedef __attribute__((ext_vector_type(8))) int          i32x8;

// TDM: move one 128-row x 64-byte tile (128x32 bf16) global -> LDS, inserting a 16B pad
// after every 64B row so the LDS row stride is 80B (== LDT elements), matching the
// bank-conflict-free ds_read_b128 fragment layout.
//   data_size      = 3 (8 bytes)
//   tile_dim0      = 8   (8 x 8B  = 64B per row)
//   tile_dim1      = 128 (rows)
//   tensor_dim0    = K_*2/8 = 1024 units available per row
//   tensor_dim1    = 128 rows available from tile start (tiles never go OOB)
//   dim0 stride    = K_*2/8 = 1024 units (8192B row pitch)
//   pad_interval=3 (16 DWORDs = 64B), pad_amount=3 (4 DWORDs = 16B)
__device__ __forceinline__ void tdm_load_tile(const __bf16* gptr, unsigned lds_byte_off) {
  unsigned long long ga = (unsigned long long)(uintptr_t)gptr;
  u32x4 g0;
  g0[0] = 1u;                                                  // count=1, gather off
  g0[1] = lds_byte_off;                                        // lds_addr
  g0[2] = (unsigned)ga;                                        // global_addr[31:0]
  g0[3] = (unsigned)((ga >> 32) & 0x01FFFFFFu) | (2u << 30);   // global_addr[56:32], type=2
  i32x8 g1;
  g1[0] = (3 << 16) | (1 << 20) | (3 << 22) | (3 << 25);       // data_size=8B, pad 16B/64B
  g1[1] = (int)(1024u << 16);                                  // tensor_dim0 = 1024 (lo16<<16)
  g1[2] = (int)(128u << 16);                                   // tensor_dim0 hi=0 | tensor_dim1=128
  g1[3] = (int)(8u << 16);                                     // tensor_dim1 hi=0 | tile_dim0=8
  g1[4] = 128;                                                 // tile_dim1=128, tile_dim2=0
  g1[5] = 1024;                                                // tensor_dim0_stride lo32
  g1[6] = 0;                                                   // stride hi | dim1_stride lo
  g1[7] = 0;
  i32x4 z4 = {};
#if defined(__clang_major__) && (__clang_major__ >= 23)
  i32x8 z8 = {};
  __builtin_amdgcn_tensor_load_to_lds(g0, g1, z4, z4, z8, 0);
#else
  __builtin_amdgcn_tensor_load_to_lds(g0, g1, z4, z4, 0);
#endif
}
#endif  // CDNA5_HAS_TDM

// ---------------- phase 2: C[B,O] = A[B,K] (bf16) x Bmat[O,K]^T (bf16), f32 accumulate -------
__global__ __launch_bounds__(256)
void wmma_gemm_kernel(const __bf16* __restrict__ A, const __bf16* __restrict__ Bm,
                      float* __restrict__ C) {
  __shared__ __align__(16) __bf16 sA[2][BM * LDT];   // 2 x 10 KB
  __shared__ __align__(16) __bf16 sB[2][BN * LDT];   // 2 x 10 KB

  const int tid  = threadIdx.x;
  const int lane = tid & 31;
  const int wid  = tid >> 5;           // 8 waves
  const int wm   = wid >> 2;           // 0..1 : wave M position (64 rows each)
  const int wn   = wid & 3;            // 0..3 : wave N position (32 cols each)
  const int hf   = lane >> 4;          // half-wave selector
  const int r16  = lane & 15;

  const int m0 = blockIdx.x * BM;
  const int n0 = blockIdx.y * BN;

  const v8f zero = {};
  v8f acc[4][2];
#pragma unroll
  for (int mt = 0; mt < 4; ++mt)
#pragma unroll
    for (int nt = 0; nt < 2; ++nt) acc[mt][nt] = zero;

#if CDNA5_HAS_TDM
  const unsigned ldsAoff[2] = {(unsigned)(uintptr_t)&sA[0][0], (unsigned)(uintptr_t)&sA[1][0]};
  const unsigned ldsBoff[2] = {(unsigned)(uintptr_t)&sB[0][0], (unsigned)(uintptr_t)&sB[1][0]};
  const __bf16* gA = A  + (size_t)m0 * K_;
  const __bf16* gB = Bm + (size_t)n0 * K_;

  // Prologue: TDM stage 0 (wave 0 issues; TENSORcnt is per-wave, barrier publishes)
  if (wid == 0) {
    tdm_load_tile(gA, ldsAoff[0]);
    tdm_load_tile(gB, ldsBoff[0]);
    __builtin_amdgcn_s_wait_tensorcnt(0);
  }
  __syncthreads();
#else
  // Fallback staging: 128 rows x 4 chunks(16B) = 512 chunks; 256 threads -> 2 chunks each.
  const int c0 = tid, c1 = tid + 256;
  const int ar0 = c0 >> 2, ac0 = (c0 & 3) * 8;
  const int ar1 = c1 >> 2, ac1 = (c1 & 3) * 8;
  const __bf16* gA0 = A  + (size_t)(m0 + ar0) * K_ + ac0;
  const __bf16* gA1 = A  + (size_t)(m0 + ar1) * K_ + ac1;
  const __bf16* gB0 = Bm + (size_t)(n0 + ar0) * K_ + ac0;
  const __bf16* gB1 = Bm + (size_t)(n0 + ar1) * K_ + ac1;
  {
    v8bf a0 = *(const v8bf*)gA0;
    v8bf a1 = *(const v8bf*)gA1;
    v8bf b0 = *(const v8bf*)gB0;
    v8bf b1 = *(const v8bf*)gB1;
    *(v8bf*)&sA[0][ar0 * LDT + ac0] = a0;
    *(v8bf*)&sA[0][ar1 * LDT + ac1] = a1;
    *(v8bf*)&sB[0][ar0 * LDT + ac0] = b0;
    *(v8bf*)&sB[0][ar1 * LDT + ac1] = b1;
  }
  __syncthreads();
#endif

#pragma unroll 1
  for (int kt = 0; kt < NKT; ++kt) {
    const int cur = kt & 1;

#if CDNA5_HAS_TDM
    // Issue stage kt+1 DMA before compute so the TDM overlaps the WMMAs.
    if (wid == 0 && kt + 1 < NKT) {
      const size_t off = (size_t)(kt + 1) * BK;
      tdm_load_tile(gA + off, ldsAoff[cur ^ 1]);
      tdm_load_tile(gB + off, ldsBoff[cur ^ 1]);
    }
#else
    v8bf a0 = {}, a1 = {}, b0 = {}, b1 = {};
    if (kt + 1 < NKT) {
      const size_t off = (size_t)(kt + 1) * BK;
      a0 = *(const v8bf*)(gA0 + off);
      a1 = *(const v8bf*)(gA1 + off);
      b0 = *(const v8bf*)(gB0 + off);
      b1 = *(const v8bf*)(gB1 + off);
    }
    if (kt + 2 < NKT) {
      __builtin_prefetch(gA0 + (size_t)(kt + 2) * BK, 0, 3);
      __builtin_prefetch(gB0 + (size_t)(kt + 2) * BK, 0, 3);
    }
#endif

    // A fragments: 16-bit A 16x32 layout -> per lane K chunks [hf*8..+7] and [16+hf*8..+7]
    v16bf afrag[4];
#pragma unroll
    for (int mt = 0; mt < 4; ++mt) {
      const __bf16* p = &sA[cur][(wm * 64 + mt * 16 + r16) * LDT];
      union { v16bf v; struct { v8bf lo, hi; } s; } u;
      u.s.lo = *(const v8bf*)(p + hf * 8);
      u.s.hi = *(const v8bf*)(p + 16 + hf * 8);
      afrag[mt] = u.v;
    }
    // B fragments: 16-bit B 32x16 layout -> per lane K chunks [hf*8..+7] and [16+hf*8..+7]
    v16bf bfrag[2];
#pragma unroll
    for (int nt = 0; nt < 2; ++nt) {
      const __bf16* p = &sB[cur][(wn * 32 + nt * 16 + r16) * LDT];
      union { v16bf v; struct { v8bf lo, hi; } s; } u;
      u.s.lo = *(const v8bf*)(p + hf * 8);
      u.s.hi = *(const v8bf*)(p + 16 + hf * 8);
      bfrag[nt] = u.v;
    }

#pragma unroll
    for (int mt = 0; mt < 4; ++mt)
#pragma unroll
      for (int nt = 0; nt < 2; ++nt)
        acc[mt][nt] = __builtin_amdgcn_wmma_f32_16x16x32_bf16(
            false, afrag[mt], false, bfrag[nt], (short)0, acc[mt][nt], false, false);

#if CDNA5_HAS_TDM
    if (wid == 0 && kt + 1 < NKT) __builtin_amdgcn_s_wait_tensorcnt(0);
#else
    if (kt + 1 < NKT) {
      const int nxt = cur ^ 1;
      *(v8bf*)&sA[nxt][ar0 * LDT + ac0] = a0;
      *(v8bf*)&sA[nxt][ar1 * LDT + ac1] = a1;
      *(v8bf*)&sB[nxt][ar0 * LDT + ac0] = b0;
      *(v8bf*)&sB[nxt][ar1 * LDT + ac1] = b1;
    }
#endif
    __syncthreads();
  }

  // Epilogue: C/D layout: lane (hf,r16): n = r16, rows m = hf*8 + r for VGPR r
#pragma unroll
  for (int mt = 0; mt < 4; ++mt)
#pragma unroll
    for (int nt = 0; nt < 2; ++nt) {
      const int m = m0 + wm * 64 + mt * 16 + hf * 8;
      const int n = n0 + wn * 32 + nt * 16 + r16;
#pragma unroll
      for (int r = 0; r < 8; ++r) C[(size_t)(m + r) * O_ + n] = acc[mt][nt][r];
    }
}

extern "C" void kernel_launch(void* const* d_in, const int* in_sizes, int n_in,
                              void* d_out, int out_size, void* d_ws, size_t ws_size,
                              hipStream_t stream) {
  (void)in_sizes; (void)n_in; (void)out_size; (void)ws_size;
  const float* x    = (const float*)d_in[0];   // [B, I] f32
  const float* coef = (const float*)d_in[1];   // [O, I, D] f32 == [O, K] row-major
  float* out = (float*)d_out;                  // [B, O] f32

  // Workspace: coef bf16 (O*K*2 = 4 MB) then jac bf16 (B*K*2 = 32 MB) -> 36 MB total
  __bf16* coefB = (__bf16*)d_ws;
  __bf16* jacB  = (__bf16*)((char*)d_ws + (size_t)O_ * K_ * sizeof(__bf16));

  coef_to_bf16_kernel<<<(O_ * K_ / 8) / 256, 256, 0, stream>>>(coef, coefB);
  jacobi_bf16_kernel<<<(B_ * I_) / 256, 256, 0, stream>>>(x, jacB);

  dim3 grid(B_ / BM, O_ / BN);                 // 32 x 4 workgroups
  wmma_gemm_kernel<<<grid, 256, 0, stream>>>(jacB, coefB, out);
}
